// RGN_63479616635468
// MI455X (gfx1250) — matrix operand order repaired and checked by reading, not compiled, not run on hardware
//
#include <hip/hip_runtime.h>
#include <hip/hip_bf16.h>
#include <math.h>

// ---------------- Problem constants ----------------
#define D_IN   42
#define HID    800
#define GATES  3200      // 4*HID
#define SEQL   256
#define BATCH  16
#define ALPHA  20
#define H2     1600      // 2*HID
#define ROWS   4096      // SEQL*BATCH

// ---------------- WMMA types ----------------
typedef __attribute__((ext_vector_type(16))) __bf16 v16bf;
typedef __attribute__((ext_vector_type(8)))  float  v8f;

union FragAB {
    v16bf v;
    unsigned int d[8];
    unsigned short u[16];
};

__device__ __forceinline__ unsigned short f2bf(float f) {
    unsigned int u = __float_as_uint(f);
    unsigned int r = (u + 0x7FFFu + ((u >> 16) & 1u)) >> 16;
    return (unsigned short)r;
}
__device__ __forceinline__ float sigm(float x) {
    return 1.0f / (1.0f + __expf(-x));
}

// A fragment: 16x32 bf16, row-major source [*, ldk]. Lane l: M=l%16.
// lanes 0-15 hold K = k0+{0..7, 16..23}; lanes 16-31 hold K = k0+{8..15, 24..31}.
__device__ __forceinline__ v16bf load_fragA(const unsigned short* __restrict__ base,
                                            int row, int ldk, int k0, int lo, int hi) {
    FragAB fa;
    const unsigned short* ap = base + (size_t)row * ldk + k0 + hi * 8;
#pragma unroll
    for (int j = 0; j < 4; ++j)
        fa.d[j] = *(const unsigned int*)(ap + 2 * j);
#pragma unroll
    for (int j = 0; j < 4; ++j)
        fa.d[4 + j] = *(const unsigned int*)(ap + 16 + 2 * j);
    return fa.v;
}

// B fragment: 32x16 bf16 (K x N) taken from weight row-major [N, K] (i.e. B^T).
// Lane l supplies column N=l%16: K = k0+{0..7, 16..23} (lanes 0-15) or k0+{8..15, 24..31} (lanes 16-31).
__device__ __forceinline__ v16bf load_fragB(const unsigned short* __restrict__ wrow,
                                            int k0, int hi) {
    FragAB fb;
    const uint4* wp = (const uint4*)(wrow + k0 + hi * 8);
    uint4 p0 = wp[0];
    uint4 p1 = wp[2];
    fb.d[0] = p0.x; fb.d[1] = p0.y; fb.d[2] = p0.z; fb.d[3] = p0.w;
    fb.d[4] = p1.x; fb.d[5] = p1.y; fb.d[6] = p1.z; fb.d[7] = p1.w;
    return fb.v;
}

// ---------------- Utility kernels ----------------
__global__ void cvt_bf16_kernel(const float* __restrict__ in,
                                unsigned short* __restrict__ out, int n) {
    int i = blockIdx.x * blockDim.x + threadIdx.x;
    if (i < n) out[i] = f2bf(in[i]);
}

__global__ void zero_u32_kernel(unsigned int* __restrict__ p, int n) {
    int i = blockIdx.x * blockDim.x + threadIdx.x;
    if (i < n) p[i] = 0u;
}

// ---------------- Layer-0 x-projection (K=42, scalar f32) ----------------
// xp0[dir][t][b][c] = b_l0[dir][c] + sum_k x[b][t][k] * w_ih0[dir][c][k]
__global__ void xproj0_kernel(const float* __restrict__ x,
                              const float* __restrict__ w_ih0,
                              const float* __restrict__ b0,
                              float* __restrict__ xp0) {
    long long idx = (long long)blockIdx.x * blockDim.x + threadIdx.x;
    const long long total = 2LL * SEQL * BATCH * GATES;
    if (idx >= total) return;
    int c   = (int)(idx % GATES);
    int b   = (int)((idx / GATES) % BATCH);
    int t   = (int)((idx / ((long long)GATES * BATCH)) % SEQL);
    int dir = (int)(idx / ((long long)GATES * BATCH * SEQL));
    const float* xr = x + ((size_t)b * SEQL + t) * D_IN;
    const float* wr = w_ih0 + ((size_t)dir * GATES + c) * D_IN;
    float s = b0[dir * GATES + c];
#pragma unroll
    for (int k = 0; k < D_IN; ++k) s += xr[k] * wr[k];
    xp0[idx] = s;
}

// ---------------- Recurrent LSTM step (WMMA bf16) ----------------
// grid = (HID/16 = 50, 2 dirs); block = 32 (one wave).
// Computes h_t @ W_hh^T for 4 gate tiles, adds precomputed x-projection,
// applies gate nonlinearities, updates c/h state, stores layer output.
__global__ __launch_bounds__(32) void lstm_step_kernel(
        const unsigned short* __restrict__ whh,   // [2, GATES, HID] bf16
        const float* __restrict__ xproj,          // [2, SEQL, BATCH, GATES]
        const unsigned short* __restrict__ hprev, // [2, BATCH, HID] bf16
        unsigned short* __restrict__ hnext,       // [2, BATCH, HID] bf16
        float* __restrict__ cstate,               // [2, BATCH, HID]
        float* __restrict__ hseq,                 // [SEQL, BATCH, 2*HID] f32
        unsigned short* __restrict__ hseq_bf,     // [SEQL, BATCH, 2*HID] bf16
        int s) {
    const int lane = threadIdx.x;
    const int lo = lane & 15;
    const int hi = lane >> 4;
    const int n0 = blockIdx.x;        // h-column tile 0..49
    const int dir = blockIdx.y;
    const int t = dir ? (SEQL - 1 - s) : s;

    v8f acc[4];
#pragma unroll
    for (int q = 0; q < 4; ++q) acc[q] = (v8f){0.f,0.f,0.f,0.f,0.f,0.f,0.f,0.f};

    const unsigned short* hp = hprev + (size_t)dir * BATCH * HID;
    const unsigned short* wd = whh   + (size_t)dir * GATES * HID;

    for (int k0 = 0; k0 < HID; k0 += 32) {
        v16bf fa = load_fragA(hp, lo, HID, k0, lo, hi);
#pragma unroll
        for (int q = 0; q < 4; ++q) {
            const unsigned short* wrow = wd + (size_t)(q * HID + n0 * 16 + lo) * HID;
            v16bf fb = load_fragB(wrow, k0, hi);
            acc[q] = __builtin_amdgcn_wmma_f32_16x16x32_bf16(
                false, fa, false, fb, (short)0, acc[q], false, false);
        }
    }

    const int n = n0 * 16 + lo;
#pragma unroll
    for (int r = 0; r < 8; ++r) {
        int m = r + hi * 8;   // batch index
        const float* xp = xproj + ((((size_t)dir * SEQL + t) * BATCH + m) * GATES);
        float gi = acc[0][r] + xp[0 * HID + n];
        float gf = acc[1][r] + xp[1 * HID + n];
        float gg = acc[2][r] + xp[2 * HID + n];
        float go = acc[3][r] + xp[3 * HID + n];
        float iv = sigm(gi), fv = sigm(gf), gv = tanhf(gg), ov = sigm(go);
        size_t sidx = ((size_t)dir * BATCH + m) * HID + n;
        float c = fv * cstate[sidx] + iv * gv;
        cstate[sidx] = c;
        float h = ov * tanhf(c);
        hnext[sidx] = f2bf(h);
        size_t oidx = (((size_t)t * BATCH + m) * H2) + (size_t)dir * HID + n;
        hseq[oidx] = h;
        hseq_bf[oidx] = f2bf(h);
    }
}

// ---------------- Layer-1 x-projection: big WMMA GEMM ----------------
// out[dir][row][c] = bias[dir][c] + A[row,:1600] @ W[dir][c,:1600]^T
// grid = (ROWS/16 = 256, GATES/64 = 50, 2); block = 32 (one wave, 4 N-tiles).
__global__ __launch_bounds__(32) void xproj1_gemm_kernel(
        const unsigned short* __restrict__ A,    // [ROWS, H2] bf16
        const unsigned short* __restrict__ W,    // [2, GATES, H2] bf16
        const float* __restrict__ bias,          // [2, GATES]
        float* __restrict__ out) {               // [2, ROWS, GATES]
    const int lane = threadIdx.x;
    const int lo = lane & 15;
    const int hi = lane >> 4;
    const int mt = blockIdx.x;
    const int ng = blockIdx.y;
    const int dir = blockIdx.z;

    v8f acc[4];
#pragma unroll
    for (int q = 0; q < 4; ++q) acc[q] = (v8f){0.f,0.f,0.f,0.f,0.f,0.f,0.f,0.f};

    const unsigned short* wd = W + (size_t)dir * GATES * H2;
    const int arow = mt * 16 + lo;

    for (int k0 = 0; k0 < H2; k0 += 32) {
        v16bf fa = load_fragA(A, arow, H2, k0, lo, hi);
#pragma unroll
        for (int q = 0; q < 4; ++q) {
            const unsigned short* wrow = wd + (size_t)(ng * 64 + q * 16 + lo) * H2;
            v16bf fb = load_fragB(wrow, k0, hi);
            acc[q] = __builtin_amdgcn_wmma_f32_16x16x32_bf16(
                false, fa, false, fb, (short)0, acc[q], false, false);
        }
    }

#pragma unroll
    for (int r = 0; r < 8; ++r) {
        int m = r + hi * 8;
        int mrow = mt * 16 + m;
#pragma unroll
        for (int q = 0; q < 4; ++q) {
            int col = ng * 64 + q * 16 + lo;
            out[((size_t)dir * ROWS + mrow) * GATES + col] =
                acc[q][r] + bias[dir * GATES + col];
        }
    }
}

// ---------------- Head: logits ----------------
__global__ void logits_kernel(const float* __restrict__ h2seq,  // [ROWS, H2]
                              const float* __restrict__ w_lin,  // [ALPHA, H2]
                              const float* __restrict__ b_lin,  // [ALPHA]
                              float* __restrict__ logits) {     // [ROWS, ALPHA]
    int idx = blockIdx.x * blockDim.x + threadIdx.x;
    if (idx >= ROWS * ALPHA) return;
    int a = idx % ALPHA;
    int row = idx / ALPHA;
    const float* hr = h2seq + (size_t)row * H2;
    const float* wr = w_lin + (size_t)a * H2;
    float s = b_lin[a];
    for (int k = 0; k < H2; ++k) s += hr[k] * wr[k];
    logits[idx] = s;
}

// ---------------- Head: softmax + atan2 over alphabet ----------------
__global__ void angles_kernel(const float* __restrict__ logits,   // [ROWS, ALPHA]
                              const float* __restrict__ alphabet, // [ALPHA, 3]
                              float* __restrict__ ang) {          // [ROWS, 3]
    int row = blockIdx.x * blockDim.x + threadIdx.x;
    if (row >= ROWS) return;
    const float* lg = logits + (size_t)row * ALPHA;
    float mx = lg[0];
#pragma unroll
    for (int a = 1; a < ALPHA; ++a) mx = fmaxf(mx, lg[a]);
    float p[ALPHA];
    float den = 0.f;
#pragma unroll
    for (int a = 0; a < ALPHA; ++a) { p[a] = __expf(lg[a] - mx); den += p[a]; }
    float inv = 1.0f / den;
#pragma unroll
    for (int k = 0; k < 3; ++k) {
        float ss = 0.f, cc = 0.f;
#pragma unroll
        for (int a = 0; a < ALPHA; ++a) {
            float w = p[a] * inv;
            float al = alphabet[a * 3 + k];
            ss += w * __sinf(al);
            cc += w * __cosf(al);
        }
        ang[row * 3 + k] = atan2f(ss, cc);
    }
}

// ---------------- Sequential coordinate extension (NeRF) ----------------
// out shape: [3 + 3*SEQL, BATCH, 3]. One lane per batch chain.
__global__ __launch_bounds__(32) void coords_kernel(const float* __restrict__ ang,
                                                    float* __restrict__ out) {
    int b = threadIdx.x;
    if (b >= BATCH) return;
    const float BL[3] = {145.801f, 152.326f, 132.868f};
    const float BA[3] = {2.124f, 1.941f, 2.028f};

    float Ax = 0.f, Ay = 0.f, Az = 1.f;
    float Bx = 0.f, By = 1.f, Bz = 0.f;
    float Cx = 1.f, Cy = 0.f, Cz = 0.f;

    // init rows
    out[(0 * BATCH + b) * 3 + 0] = Ax; out[(0 * BATCH + b) * 3 + 1] = Ay; out[(0 * BATCH + b) * 3 + 2] = Az;
    out[(1 * BATCH + b) * 3 + 0] = Bx; out[(1 * BATCH + b) * 3 + 1] = By; out[(1 * BATCH + b) * 3 + 2] = Bz;
    out[(2 * BATCH + b) * 3 + 0] = Cx; out[(2 * BATCH + b) * 3 + 1] = Cy; out[(2 * BATCH + b) * 3 + 2] = Cz;

    for (int t = 0; t < SEQL; ++t) {
        for (int i = 0; i < 3; ++i) {
            float T = BA[i], R = BL[i];
            float P = ang[((size_t)t * BATCH + b) * 3 + i];
            float sT = sinf(T), cT = cosf(T);
            float d2x = -R * cT;
            float d2y = R * cosf(P) * sT;
            float d2z = R * sinf(P) * sT;
            // bc = normalize(C - B)
            float bx = Cx - Bx, by = Cy - By, bz = Cz - Bz;
            float rn = rsqrtf(bx * bx + by * by + bz * bz);
            bx *= rn; by *= rn; bz *= rn;
            // n = normalize(cross(B - A, bc))
            float ux = Bx - Ax, uy = By - Ay, uz = Bz - Az;
            float nx = uy * bz - uz * by;
            float ny = uz * bx - ux * bz;
            float nz = ux * by - uy * bx;
            float rn2 = rsqrtf(nx * nx + ny * ny + nz * nz);
            nx *= rn2; ny *= rn2; nz *= rn2;
            // m2 = cross(n, bc)
            float mx = ny * bz - nz * by;
            float my = nz * bx - nx * bz;
            float mz = nx * by - ny * bx;
            float Dx = bx * d2x + mx * d2y + nx * d2z + Cx;
            float Dy = by * d2x + my * d2y + ny * d2z + Cy;
            float Dz = bz * d2x + mz * d2y + nz * d2z + Cz;
            size_t row = 3 + (size_t)t * 3 + i;
            out[(row * BATCH + b) * 3 + 0] = Dx;
            out[(row * BATCH + b) * 3 + 1] = Dy;
            out[(row * BATCH + b) * 3 + 2] = Dz;
            Ax = Bx; Ay = By; Az = Bz;
            Bx = Cx; By = Cy; Bz = Cz;
            Cx = Dx; Cy = Dy; Cz = Dz;
        }
    }
}

// ---------------- Host orchestration ----------------
extern "C" void kernel_launch(void* const* d_in, const int* in_sizes, int n_in,
                              void* d_out, int out_size, void* d_ws, size_t ws_size,
                              hipStream_t stream) {
    const float* x        = (const float*)d_in[0];  // [B, L, D_IN]
    const float* w_ih_l0  = (const float*)d_in[1];  // [2, 3200, 42]
    const float* w_hh_l0  = (const float*)d_in[2];  // [2, 3200, 800]
    const float* b_l0     = (const float*)d_in[3];  // [2, 3200]
    const float* w_ih_l1  = (const float*)d_in[4];  // [2, 3200, 1600]
    const float* w_hh_l1  = (const float*)d_in[5];  // [2, 3200, 800]
    const float* b_l1     = (const float*)d_in[6];  // [2, 3200]
    const float* w_lin    = (const float*)d_in[7];  // [20, 1600]
    const float* b_lin    = (const float*)d_in[8];  // [20]
    const float* alphabet = (const float*)d_in[9];  // [20, 3]
    float* out = (float*)d_out;

    // ---- carve workspace ----
    char* base = (char*)d_ws;
    size_t off = 0;
    auto carve = [&](size_t bytes) -> char* {
        char* p = base + off;
        off = (off + bytes + 255) & ~(size_t)255;
        return p;
    };
    const size_t N_WH  = 2ull * GATES * HID;    // w_hh elements
    const size_t N_WI1 = 2ull * GATES * H2;     // w_ih_l1 elements
    const size_t N_XP  = 2ull * SEQL * BATCH * GATES;
    const size_t N_HSQ = (size_t)SEQL * BATCH * H2;
    const size_t N_HBF = 2ull * BATCH * HID;

    unsigned short* wh0b  = (unsigned short*)carve(N_WH  * 2);
    unsigned short* wh1b  = (unsigned short*)carve(N_WH  * 2);
    unsigned short* wi1b  = (unsigned short*)carve(N_WI1 * 2);
    float*          xp0   = (float*)         carve(N_XP  * 4);
    float*          xp1   = (float*)         carve(N_XP  * 4);
    float*          h1seq = (float*)         carve(N_HSQ * 4);
    unsigned short* h1bf  = (unsigned short*)carve(N_HSQ * 2);
    float*          h2seq = (float*)         carve(N_HSQ * 4);
    unsigned short* h2bf  = (unsigned short*)carve(N_HSQ * 2);  // sink for layer-1
    unsigned short* hbufA = (unsigned short*)carve(N_HBF * 2);
    unsigned short* hbufB = (unsigned short*)carve(N_HBF * 2);
    float*          cst   = (float*)         carve(N_HBF * 4);
    float*          logit = (float*)         carve((size_t)ROWS * ALPHA * 4);
    float*          angb  = (float*)         carve((size_t)ROWS * 3 * 4);
    (void)ws_size; (void)in_sizes; (void)n_in; (void)out_size;

    const int TB = 256;
    auto blocks = [&](long long n) { return (unsigned)((n + TB - 1) / TB); };

    // ---- weight conversion f32 -> bf16 ----
    cvt_bf16_kernel<<<blocks(N_WH),  TB, 0, stream>>>(w_hh_l0, wh0b, (int)N_WH);
    cvt_bf16_kernel<<<blocks(N_WH),  TB, 0, stream>>>(w_hh_l1, wh1b, (int)N_WH);
    cvt_bf16_kernel<<<blocks(N_WI1), TB, 0, stream>>>(w_ih_l1, wi1b, (int)N_WI1);

    // ---- layer-0 x-projection ----
    xproj0_kernel<<<blocks((long long)N_XP), TB, 0, stream>>>(x, w_ih_l0, b_l0, xp0);

    // ---- layer 0 recurrence ----
    zero_u32_kernel<<<blocks(N_HBF / 2), TB, 0, stream>>>((unsigned int*)hbufA, (int)(N_HBF / 2));
    zero_u32_kernel<<<blocks(N_HBF),     TB, 0, stream>>>((unsigned int*)cst,   (int)N_HBF);
    {
        dim3 grid(HID / 16, 2);
        for (int s = 0; s < SEQL; ++s) {
            const unsigned short* hp = (s & 1) ? hbufB : hbufA;
            unsigned short*       hn = (s & 1) ? hbufA : hbufB;
            lstm_step_kernel<<<grid, 32, 0, stream>>>(wh0b, xp0, hp, hn, cst,
                                                      h1seq, h1bf, s);
        }
    }

    // ---- layer-1 x-projection: big WMMA GEMM ----
    {
        dim3 grid(ROWS / 16, GATES / 64, 2);
        xproj1_gemm_kernel<<<grid, 32, 0, stream>>>(h1bf, wi1b, b_l1, xp1);
    }

    // ---- layer 1 recurrence ----
    zero_u32_kernel<<<blocks(N_HBF / 2), TB, 0, stream>>>((unsigned int*)hbufA, (int)(N_HBF / 2));
    zero_u32_kernel<<<blocks(N_HBF),     TB, 0, stream>>>((unsigned int*)cst,   (int)N_HBF);
    {
        dim3 grid(HID / 16, 2);
        for (int s = 0; s < SEQL; ++s) {
            const unsigned short* hp = (s & 1) ? hbufB : hbufA;
            unsigned short*       hn = (s & 1) ? hbufA : hbufB;
            lstm_step_kernel<<<grid, 32, 0, stream>>>(wh1b, xp1, hp, hn, cst,
                                                      h2seq, h2bf, s);
        }
    }

    // ---- head ----
    logits_kernel<<<blocks((long long)ROWS * ALPHA), TB, 0, stream>>>(h2seq, w_lin, b_lin, logit);
    angles_kernel<<<blocks(ROWS), TB, 0, stream>>>(logit, alphabet, angb);

    // ---- sequential coordinate chain ----
    coords_kernel<<<1, 32, 0, stream>>>(angb, out);
}
